// FeaturePropagation_60833916780805
// MI455X (gfx1250) — matrix operand order repaired and checked by reading, not compiled, not run on hardware
//
#include <hip/hip_runtime.h>
#include <hip/hip_bf16.h>
#include <stdint.h>

// Problem constants (from reference)
#define BQ   2
#define NQ   32768
#define MQ   4096
#define CQ   128
#define OUTC 128
#define INC  131      // 3 + C
#define KP   160      // K padded to 5 tiles of 32 for GEMM1
#define NTOT (BQ * NQ)

// h layout per row (KP f16): [0..127]=interp features, [128..130]=xyz, [131..159]=0
// Weight repack compensates: h-col k<128 -> ref col 3+k ; 128<=k<131 -> ref col k-128

typedef __attribute__((ext_vector_type(16))) _Float16 v16h;
typedef __attribute__((ext_vector_type(8)))  float    v8f;
typedef __attribute__((ext_vector_type(8)))  uint32_t v8u;

// A-fragment K base for v_wmma_f32_16x16x32_f16 (16x32 A, lane half = lane>>4)
__device__ __forceinline__ int a_k0(int j, int half) {
    return (j < 4) ? (8 * half + 2 * j) : (16 + 8 * half + 2 * (j - 4));
}

// ---------------------------------------------------------------------------
// Kernel 0: repack w1 (with K-permutation + padding) and w2 into per-lane
// WMMA B-fragment order, f16 pairs packed in uint32.
// w1p: 5 kt * 8 nt frags, each 32 lanes * 8 dwords  -> 10240 dwords
// w2p: 4 kt * 8 nt frags                            ->  8192 dwords
// ---------------------------------------------------------------------------
__global__ __launch_bounds__(256) void repack_weights_kernel(
    const float* __restrict__ w1, const float* __restrict__ w2,
    uint32_t* __restrict__ w1p, uint32_t* __restrict__ w2p)
{
    int id = blockIdx.x * 256 + threadIdx.x;
    if (id < 10240) {
        int f = id >> 8, rem = id & 255, lane = rem >> 3, j = rem & 7;
        int kt = f >> 3, nt = f & 7;
        int n  = nt * 16 + (lane & 15);
        int kb = kt * 32 + a_k0(j, lane >> 4);
        uint32_t packed = 0;
        for (int p = 0; p < 2; ++p) {
            int k = kb + p;
            float v = 0.0f;
            if (k < 128)      v = w1[n * INC + 3 + k];       // interp channels
            else if (k < 131) v = w1[n * INC + (k - 128)];   // xyz coords
            _Float16 hv = (_Float16)v;
            packed |= ((uint32_t)__builtin_bit_cast(unsigned short, hv)) << (16 * p);
        }
        w1p[id] = packed;
    }
    int id2 = id - 10240;
    if (id2 >= 0 && id2 < 8192) {
        int f = id2 >> 8, rem = id2 & 255, lane = rem >> 3, j = rem & 7;
        int kt = f >> 3, nt = f & 7;
        int n  = nt * 16 + (lane & 15);
        int kb = kt * 32 + a_k0(j, lane >> 4);
        uint32_t packed = 0;
        for (int p = 0; p < 2; ++p) {
            _Float16 hv = (_Float16)w2[n * OUTC + kb + p];
            packed |= ((uint32_t)__builtin_bit_cast(unsigned short, hv)) << (16 * p);
        }
        w2p[id2] = packed;
    }
}

// ---------------------------------------------------------------------------
// Kernel 1: brute-force 3-NN. Candidates LDS-resident AoS float4(x,y,z,|s|^2);
// rank by score = |s|^2 - 2 x.s (same order as squared distance). Groups of 8
// are screened with an fminf tree; the top-3 insertion chain runs under a
// scalar branch taken only when the group can beat the current third-best
// (rare after warm-up) -> ~5 issue slots per candidate on the fast path.
// Then gather + average neighbor features, emit f16 h rows [NTOT x KP].
// ---------------------------------------------------------------------------
__global__ __launch_bounds__(256) void knn_interp_kernel(
    const float* __restrict__ x, const float* __restrict__ sxyz,
    const float* __restrict__ feat, _Float16* __restrict__ h)
{
    __shared__ float4 sp[MQ];                    // 64 KB (WGP has 320 KB)
    const int tid = threadIdx.x;
    const int b   = blockIdx.y;

    const float* sb = sxyz + (size_t)b * MQ * 3;
    for (int i = tid; i < MQ; i += 256) {
        float a0 = sb[i * 3 + 0];
        float a1 = sb[i * 3 + 1];
        float a2 = sb[i * 3 + 2];
        sp[i] = make_float4(a0, a1, a2, fmaf(a0, a0, fmaf(a1, a1, a2 * a2)));
    }
    __syncthreads();

    const int n = blockIdx.x * 256 + tid;
    const float* xp = x + ((size_t)b * NQ + n) * 3;
    const float px = xp[0], py = xp[1], pz = xp[2];
    const float nx = -2.0f * px, ny = -2.0f * py, nz = -2.0f * pz;

    float d0 = 3.4e38f, d1 = 3.4e38f, d2v = 3.4e38f;
    int   i0 = 0, i1 = 0, i2 = 0;

    for (int mg = 0; mg < MQ; mg += 8) {
        float dsc[8];
#pragma unroll
        for (int t = 0; t < 8; ++t) {
            float4 s = sp[mg + t];
            dsc[t] = fmaf(s.x, nx, fmaf(s.y, ny, fmaf(s.z, nz, s.w)));
        }
        float m01 = fminf(dsc[0], dsc[1]);
        float m23 = fminf(dsc[2], dsc[3]);
        float m45 = fminf(dsc[4], dsc[5]);
        float m67 = fminf(dsc[6], dsc[7]);
        float gmin = fminf(fminf(m01, m23), fminf(m45, m67));
        if (gmin < d2v) {                         // rare: real branch skips chain
#pragma unroll
            for (int t = 0; t < 8; ++t) {
                float d = dsc[t];
                int   m = mg + t;
                if (d < d2v) {
                    if (d < d1) {
                        d2v = d1; i2 = i1;
                        if (d < d0) { d1 = d0; i1 = i0; d0 = d; i0 = m; }
                        else        { d1 = d;  i1 = m; }
                    } else { d2v = d; i2 = m; }
                }
            }
        }
    }

    const float4* F0 = (const float4*)(feat + ((size_t)b * MQ + i0) * CQ);
    const float4* F1 = (const float4*)(feat + ((size_t)b * MQ + i1) * CQ);
    const float4* F2 = (const float4*)(feat + ((size_t)b * MQ + i2) * CQ);

    const size_t row = (size_t)b * NQ + n;
    _Float16* hr = h + row * KP;                  // row base: 320 B -> 8B aligned
    const float inv3 = 1.0f / 3.0f;

#pragma unroll 4
    for (int c4 = 0; c4 < CQ / 4; ++c4) {
        float4 a = F0[c4], bb = F1[c4], cc = F2[c4];
        union { _Float16 hv[4]; uint2 u; } pk;
        pk.hv[0] = (_Float16)((a.x + bb.x + cc.x) * inv3);
        pk.hv[1] = (_Float16)((a.y + bb.y + cc.y) * inv3);
        pk.hv[2] = (_Float16)((a.z + bb.z + cc.z) * inv3);
        pk.hv[3] = (_Float16)((a.w + bb.w + cc.w) * inv3);
        ((uint2*)hr)[c4] = pk.u;
    }
    hr[128] = (_Float16)px;
    hr[129] = (_Float16)py;
    hr[130] = (_Float16)pz;
    hr[131] = (_Float16)0.0f;
    uint32_t* hz = (uint32_t*)(hr + 132);         // 28 halves of zero pad
#pragma unroll
    for (int t = 0; t < 14; ++t) hz[t] = 0u;
}

// ---------------------------------------------------------------------------
// Kernel 2: fused 2-layer MLP via WMMA. Block = 128 threads = 4 waves; each
// wave owns TWO 16-row tiles (32 rows) across all 128 output channels so each
// B fragment loaded from L2 feeds two WMMAs. GEMM1 (K=160) -> bias+ReLU ->
// LDS lane remap -> GEMM2 (K=128) -> bias -> direct [B,OUT,N] stores (each
// lane writes two runs of 8 consecutive f32 per channel -> 32B segments, L2 merges).
// ---------------------------------------------------------------------------
#define H1LD 132   // padded LDS row stride (halves) -> conflict-free ds_read_b32

__global__ __launch_bounds__(128) void mlp_wmma_kernel(
    const _Float16* __restrict__ h, const uint32_t* __restrict__ w1p,
    const uint32_t* __restrict__ w2p, const float* __restrict__ b1,
    const float* __restrict__ b2, float* __restrict__ out)
{
    __shared__ _Float16 h1s[4][32 * H1LD];        // 33.8 KB

    const int lane = threadIdx.x & 31;
    const int wv   = threadIdx.x >> 5;
    const int m_lo = lane & 15;
    const int half = lane >> 4;
    const int m_c  = 8 * half;                    // C-layout row offset
    const int rowBase = blockIdx.x * 128 + wv * 32;

    // ---------------- GEMM1: h[32x160] x W1[160x128] ----------------
    v8f acc[2][8];
#pragma unroll
    for (int mt = 0; mt < 2; ++mt)
#pragma unroll
        for (int nt = 0; nt < 8; ++nt) acc[mt][nt] = 0.0f;

    const uint32_t* hrow0 = (const uint32_t*)(h + (size_t)(rowBase + m_lo) * KP);
    const uint32_t* hrow1 = (const uint32_t*)(h + (size_t)(rowBase + 16 + m_lo) * KP);
#pragma unroll
    for (int kt = 0; kt < 5; ++kt) {
        v8u au0, au1;
#pragma unroll
        for (int j = 0; j < 8; ++j) {
            int idx = (kt * 32 + a_k0(j, half)) >> 1;
            au0[j] = hrow0[idx];
            au1[j] = hrow1[idx];
        }
        v16h a0 = __builtin_bit_cast(v16h, au0);
        v16h a1 = __builtin_bit_cast(v16h, au1);
#pragma unroll
        for (int nt = 0; nt < 8; ++nt) {
            const uint32_t* bp = w1p + (((kt * 8 + nt) * 32 + lane) << 3);
            v8u bu;
#pragma unroll
            for (int j = 0; j < 8; ++j) bu[j] = bp[j];
            v16h bf = __builtin_bit_cast(v16h, bu);
            acc[0][nt] = __builtin_amdgcn_wmma_f32_16x16x32_f16(
                false, a0, false, bf, (short)0, acc[0][nt], false, false);
            acc[1][nt] = __builtin_amdgcn_wmma_f32_16x16x32_f16(
                false, a1, false, bf, (short)0, acc[1][nt], false, false);
        }
    }

    // bias + ReLU, remap C-layout -> f16 A-layout staging in LDS
#pragma unroll
    for (int nt = 0; nt < 8; ++nt) {
        float bias = b1[nt * 16 + m_lo];
#pragma unroll
        for (int mt = 0; mt < 2; ++mt)
#pragma unroll
            for (int r = 0; r < 8; ++r) {
                float v = acc[mt][nt][r] + bias;
                v = v > 0.0f ? v : 0.0f;
                h1s[wv][(mt * 16 + r + m_c) * H1LD + (nt * 16 + m_lo)] = (_Float16)v;
            }
    }
    __syncthreads();

    // ---------------- GEMM2: h1[32x128] x W2[128x128] ----------------
    v8f acc2[2][8];
#pragma unroll
    for (int mt = 0; mt < 2; ++mt)
#pragma unroll
        for (int nt = 0; nt < 8; ++nt) acc2[mt][nt] = 0.0f;

    const uint32_t* hr20 = (const uint32_t*)&h1s[wv][m_lo * H1LD];
    const uint32_t* hr21 = (const uint32_t*)&h1s[wv][(16 + m_lo) * H1LD];
#pragma unroll
    for (int kt = 0; kt < 4; ++kt) {
        v8u au0, au1;
#pragma unroll
        for (int j = 0; j < 8; ++j) {
            int idx = (kt * 32 + a_k0(j, half)) >> 1;
            au0[j] = hr20[idx];
            au1[j] = hr21[idx];
        }
        v16h a0 = __builtin_bit_cast(v16h, au0);
        v16h a1 = __builtin_bit_cast(v16h, au1);
#pragma unroll
        for (int nt = 0; nt < 8; ++nt) {
            const uint32_t* bp = w2p + (((kt * 8 + nt) * 32 + lane) << 3);
            v8u bu;
#pragma unroll
            for (int j = 0; j < 8; ++j) bu[j] = bp[j];
            v16h bf = __builtin_bit_cast(v16h, bu);
            acc2[0][nt] = __builtin_amdgcn_wmma_f32_16x16x32_f16(
                false, a0, false, bf, (short)0, acc2[0][nt], false, false);
            acc2[1][nt] = __builtin_amdgcn_wmma_f32_16x16x32_f16(
                false, a1, false, bf, (short)0, acc2[1][nt], false, false);
        }
    }

    // bias + direct channels-first stores: lane writes two runs of 8 consecutive f32 per
    // channel (32B each), L2 write-combines before HBM writeback.
    const int bb    = rowBase >> 15;              // batch (blocks never straddle)
    const int nbase = rowBase & 32767;
#pragma unroll
    for (int nt = 0; nt < 8; ++nt) {
        float bias = b2[nt * 16 + m_lo];
        float* p = out + ((size_t)(bb * OUTC + nt * 16 + m_lo)) * NQ + nbase;
#pragma unroll
        for (int mt = 0; mt < 2; ++mt)
#pragma unroll
            for (int r = 0; r < 8; ++r)
                p[mt * 16 + r + m_c] = acc2[mt][nt][r] + bias;
    }
}

// ---------------------------------------------------------------------------
extern "C" void kernel_launch(void* const* d_in, const int* in_sizes, int n_in,
                              void* d_out, int out_size, void* d_ws, size_t ws_size,
                              hipStream_t stream) {
    const float* x    = (const float*)d_in[0];
    const float* sxyz = (const float*)d_in[1];
    const float* feat = (const float*)d_in[2];
    const float* w1   = (const float*)d_in[3];
    const float* b1   = (const float*)d_in[4];
    const float* w2   = (const float*)d_in[5];
    const float* b2   = (const float*)d_in[6];
    float* out        = (float*)d_out;

    // workspace: h [NTOT x KP] f16, then fragment-packed weights
    _Float16* h   = (_Float16*)d_ws;
    uint32_t* w1p = (uint32_t*)((char*)d_ws + (size_t)NTOT * KP * sizeof(_Float16));
    uint32_t* w2p = w1p + 10240;

    repack_weights_kernel<<<72, 256, 0, stream>>>(w1, w2, w1p, w2p);
    knn_interp_kernel<<<dim3(NQ / 256, BQ), 256, 0, stream>>>(x, sxyz, feat, h);
    mlp_wmma_kernel<<<NTOT / 128, 128, 0, stream>>>(h, w1p, w2p, b1, b2, out);
}
